// CBSA_80771154968828
// MI455X (gfx1250) — matrix operand order repaired and checked by reading, not compiled, not run on hardware
//
#include <hip/hip_runtime.h>
#include <hip/hip_bf16.h>
#include <stdint.h>

// ---------------------------------------------------------------------------
// CBSA (pooled cross/self attention) for MI455X / gfx1250, wave32 + WMMA bf16.
// All five matmul stages run on v_wmma_f32_16x16x32_bf16 (fp32 accumulate).
// The big GEMMs stage their A tiles via the Tensor Data Mover
// (TENSOR_LOAD_TO_LDS, double-buffered, TENSORcnt-tracked) so the DMA engine
// feeds LDS while the SIMDs issue WMMAs. ~104 MB workspace.
// ---------------------------------------------------------------------------

typedef unsigned short u16;
typedef __bf16 bf16t;
typedef bf16t v16bf __attribute__((ext_vector_type(16)));
typedef float v8f __attribute__((ext_vector_type(8)));
typedef unsigned int v4u __attribute__((ext_vector_type(4)));
typedef int v4i __attribute__((ext_vector_type(4)));
typedef int v8i __attribute__((ext_vector_type(8)));

#define DIMC   512
#define HEADS  8
#define DH     64
#define BATCH  8
#define NTOK   4096
#define NQ     64
#define ATTN_SCALE 0.125f

#ifndef __has_builtin
#define __has_builtin(x) 0
#endif
#if __has_builtin(__builtin_amdgcn_tensor_load_to_lds) && \
    __has_builtin(__builtin_amdgcn_s_wait_tensorcnt)
#define HAVE_TDM 1
#else
#define HAVE_TDM 0
#endif

union Frag { v16bf v; uint4 q[2]; };
union U4u  { uint4 v; u16 s[8]; };

__device__ __forceinline__ u16 f2bf(float f) {
  uint32_t u = __float_as_uint(f);
  return (u16)((u + 0x7FFFu + ((u >> 16) & 1u)) >> 16);   // RNE truncation
}
__device__ __forceinline__ float bf2f(u16 h) {
  return __uint_as_float(((uint32_t)h) << 16);
}

// A-matrix fragment, 16x32 bf16 row-major source. Per ISA 7.12.2:
// lanes 0-15 (row=lane): K 0-7 in v0-3, K 16-23 in v4-7
// lanes 16-31 (row=lane-16): K 8-15 in v0-3, K 24-31 in v4-7
__device__ __forceinline__ v16bf load_frag_a(const u16* base, int stride, int lane) {
  const int hf = lane >> 4, idx = lane & 15;
  const u16* p = base + (size_t)idx * stride + 8 * hf;
  Frag f;
  f.q[0] = *(const uint4*)(p);
  f.q[1] = *(const uint4*)(p + 16);
  return f.v;
}
// B-matrix fragment, 32x16 bf16; source laid out column-contiguous over K
// (i.e. row n of a [N][K] array). lanes 0-15: K 0-7,16-23; lanes 16-31: K 8-15,24-31.
__device__ __forceinline__ v16bf load_frag_b(const u16* base, int stride, int lane) {
  const int hf = lane >> 4, idx = lane & 15;
  const u16* p = base + (size_t)idx * stride + 8 * hf;
  Frag f;
  f.q[0] = *(const uint4*)(p);
  f.q[1] = *(const uint4*)(p + 16);
  return f.v;
}
__device__ __forceinline__ v8f wmma_bf16(v16bf a, v16bf b, v8f c) {
  return __builtin_amdgcn_wmma_f32_16x16x32_bf16(false, a, false, b, (short)0, c,
                                                 false, false);
}

#if HAVE_TDM
// Issue a TDM 2D tile load (bf16 elements): tile [tile_d1 rows x tile_d0 elems]
// from a row-major tensor with dim0-stride `stride0`, packed into LDS at
// `lds_addr`. D# layout per CDNA5 ISA 8.3/8.4 (groups 2/3 unused -> zeros).
__device__ __forceinline__ void tdm_load_2d(uint32_t lds_addr, const void* gptr,
                                            uint32_t tensor_d0, uint32_t tensor_d1,
                                            uint32_t tile_d0, uint32_t tile_d1,
                                            uint32_t stride0) {
  uint64_t ga = (uint64_t)(uintptr_t)gptr;
  v4u g0;
  g0.x = 1u;                                   // count=1, user desc, no gather
  g0.y = lds_addr;                             // LDS byte address
  g0.z = (uint32_t)ga;                         // global_addr[31:0]
  g0.w = (uint32_t)(ga >> 32) | (2u << 30);    // global_addr[56:32] | type=2
  v8i g1;
  g1[0] = (int)(1u << 16);                     // wg_mask=0, data_size=1 (2B)
  g1[1] = (int)((tensor_d0 & 0xFFFFu) << 16);  // tensor_dim0[15:0]
  g1[2] = (int)((tensor_d0 >> 16) | ((tensor_d1 & 0xFFFFu) << 16));
  g1[3] = (int)((tensor_d1 >> 16) | (tile_d0 << 16));   // tile_dim0
  g1[4] = (int)(tile_d1 & 0xFFFFu);            // tile_dim1 (tile_dim2=0)
  g1[5] = (int)stride0;                        // tensor_dim0_stride[31:0]
  g1[6] = 0;                                   // stride0 hi, stride1 lo
  g1[7] = 0;                                   // stride1 hi
  v4i z4 = {0, 0, 0, 0};
#if __clang_major__ >= 23
  v8i z8 = {0, 0, 0, 0, 0, 0, 0, 0};
  __builtin_amdgcn_tensor_load_to_lds(g0, g1, z4, z4, z8, 0);
#else
  __builtin_amdgcn_tensor_load_to_lds(g0, g1, z4, z4, 0);
#endif
}
#endif  // HAVE_TDM

// ---------------------------------------------------------------------------
// K0: fp32 -> bf16 conversion of x, proj_w, to_out_w
// ---------------------------------------------------------------------------
__global__ __launch_bounds__(256) void k_convert(
    const float* __restrict__ x, const float* __restrict__ pw,
    const float* __restrict__ tw, u16* __restrict__ x_bf,
    u16* __restrict__ pw_bf, u16* __restrict__ tw_bf) {
  size_t i = (size_t)blockIdx.x * 256 + threadIdx.x;
  const size_t NX = (size_t)BATCH * NTOK * DIMC;
  const size_t NW = (size_t)DIMC * DIMC;
  if (i < NX) {
    x_bf[i] = f2bf(x[i]);
  } else if (i < NX + NW) {
    size_t j = i - NX; pw_bf[j] = f2bf(pw[j]);
  } else if (i < NX + 2 * NW) {
    size_t j = i - NX - NW; tw_bf[j] = f2bf(tw[j]);
  }
}

// ---------------------------------------------------------------------------
// K1/K6: GEMM out[M,512] = A[M,512](bf16) @ B[512,512](bf16, row d over k)^T
// Block tile 128(M) x 64(N), 8 waves (4Mx2N), wave tile 32x32 = 2x2 frags.
// A tile staged into LDS by the Tensor Data Mover (double-buffered, wave 0
// issues, TENSORcnt-tracked); B rows served from L2 (512KB, fully resident).
// ---------------------------------------------------------------------------
__global__ __launch_bounds__(256) void k_gemm(
    const u16* __restrict__ A, const u16* __restrict__ Bw,
    const float* __restrict__ bias, u16* __restrict__ out_bf,
    float* __restrict__ out_f32) {
  __shared__ u16 As[2 * 128 * 32];
  const int tid = threadIdx.x, lane = tid & 31, wid = tid >> 5;
  const int wm = wid >> 1, wn = wid & 1;
  const int m0 = blockIdx.x * 128, n0 = blockIdx.y * 64;
  v8f acc[2][2] = {};

#if HAVE_TDM
  if (wid == 0)                 // TDM ignores EXEC; one issue per descriptor
    tdm_load_2d((uint32_t)(uintptr_t)As, A + (size_t)m0 * DIMC,
                /*tensor*/ DIMC, 128, /*tile*/ 32, 128, /*stride0*/ DIMC);
  for (int ks = 0; ks < 16; ++ks) {
    const u16* Abuf = As + (ks & 1) * (128 * 32);
    if (wid == 0) {
      if (ks + 1 < 16) {        // overlap next DMA with this step's WMMAs
        tdm_load_2d((uint32_t)(uintptr_t)(As + ((ks + 1) & 1) * (128 * 32)),
                    A + (size_t)m0 * DIMC + (ks + 1) * 32,
                    DIMC, 128, 32, 128, DIMC);
        __builtin_amdgcn_s_wait_tensorcnt(1);   // retire tile ks
      } else {
        __builtin_amdgcn_s_wait_tensorcnt(0);
      }
    }
    __syncthreads();
    v16bf bfrag[2];
#pragma unroll
    for (int ni = 0; ni < 2; ++ni)
      bfrag[ni] = load_frag_b(Bw + (size_t)(n0 + wn * 32 + ni * 16) * DIMC + ks * 32,
                              DIMC, lane);
#pragma unroll
    for (int mi = 0; mi < 2; ++mi) {
      v16bf afrag = load_frag_a(Abuf + (wm * 32 + mi * 16) * 32, 32, lane);
#pragma unroll
      for (int ni = 0; ni < 2; ++ni)
        acc[mi][ni] = wmma_bf16(afrag, bfrag[ni], acc[mi][ni]);
    }
    __syncthreads();            // protect buffer before it is refilled
  }
#else
  for (int ks = 0; ks < 16; ++ks) {
    __syncthreads();
    {
      int rl = tid >> 1, ch = (tid & 1) * 16;
      const u16* src = A + (size_t)(m0 + rl) * DIMC + ks * 32 + ch;
      uint4 a0 = *(const uint4*)(src);
      uint4 a1 = *(const uint4*)(src + 8);
      *(uint4*)(As + rl * 32 + ch)     = a0;
      *(uint4*)(As + rl * 32 + ch + 8) = a1;
      if (ks + 1 < 16) __builtin_prefetch(src + 32, 0, 1);
    }
    __syncthreads();
    v16bf bfrag[2];
#pragma unroll
    for (int ni = 0; ni < 2; ++ni)
      bfrag[ni] = load_frag_b(Bw + (size_t)(n0 + wn * 32 + ni * 16) * DIMC + ks * 32,
                              DIMC, lane);
#pragma unroll
    for (int mi = 0; mi < 2; ++mi) {
      v16bf afrag = load_frag_a(As + (wm * 32 + mi * 16) * 32, 32, lane);
#pragma unroll
      for (int ni = 0; ni < 2; ++ni)
        acc[mi][ni] = wmma_bf16(afrag, bfrag[ni], acc[mi][ni]);
    }
  }
#endif

  const int hf = lane >> 4, idx = lane & 15;
#pragma unroll
  for (int mi = 0; mi < 2; ++mi)
#pragma unroll
    for (int ni = 0; ni < 2; ++ni) {
      int col = n0 + wn * 32 + ni * 16 + idx;
#pragma unroll
      for (int r = 0; r < 8; ++r) {
        int row = m0 + wm * 32 + mi * 16 + hf * 8 + r;
        float v = acc[mi][ni][r];
        if (out_f32) out_f32[(size_t)row * DIMC + col] = v + bias[col];
        else         out_bf [(size_t)row * DIMC + col] = f2bf(v);
      }
    }
}

// ---------------------------------------------------------------------------
// K2: AdaptiveAvgPool2d((8,8)) over the 64x64 token grid (exact block mean).
// ---------------------------------------------------------------------------
__global__ __launch_bounds__(256) void k_pool(
    const u16* __restrict__ w_bf, float* __restrict__ rep_f32,
    u16* __restrict__ rep_bf) {
  int gid = blockIdx.x * 256 + threadIdx.x;         // B*64*(512/8) = 32768
  int b = gid >> 12, rem = gid & 4095;
  int q = rem >> 6, c8 = rem & 63;
  int c = c8 * 8;
  int a = q >> 3, cc = q & 7;
  float s[8] = {};
  for (int i = 0; i < 8; ++i)
    for (int j = 0; j < 8; ++j) {
      int n = (a * 8 + i) * 64 + cc * 8 + j;
      U4u u; u.v = *(const uint4*)(w_bf + ((size_t)b * NTOK + n) * DIMC + c);
#pragma unroll
      for (int e = 0; e < 8; ++e) s[e] += bf2f(u.s[e]);
    }
  size_t base = ((size_t)b * NQ + q) * DIMC + c;
  U4u o;
#pragma unroll
  for (int e = 0; e < 8; ++e) {
    float v = s[e] * (1.0f / 64.0f);
    rep_f32[base + e] = v;
    o.s[e] = f2bf(v);
  }
  *(uint4*)(rep_bf + base) = o.v;
}

// ---------------------------------------------------------------------------
// K3: cross-attention per (b,h). Q = pooled reps [64 x 64], K=V = w [4096 x 64].
// Two-pass streaming softmax (key blocks of 128):
//   pass 1: row maxima (WMMA S + lane butterflies)
//   pass 2: recompute S, P = exp(S - max), store UNNORMALIZED attn (bf16),
//           accumulate P@V via WMMA using an LDS-transposed V tile.
// rep' = rep + step_rep * (P@V)/rowsum ; rowsum exported for consumers.
// ---------------------------------------------------------------------------
__global__ __launch_bounds__(256) void k_attn1(
    const u16* __restrict__ w_bf, const u16* __restrict__ rep_bf,
    const float* __restrict__ rep_f32, const float* __restrict__ step_rep,
    u16* __restrict__ attn_bf, u16* __restrict__ reph_bf,
    float* __restrict__ rowsum_g) {
  __shared__ u16 VT[64 * 128];       // V^T tile  [d][key]
  __shared__ u16 PT[64 * 128];       // P tile    [q][key]
  __shared__ float redbuf[4][64];
  __shared__ float rowmax_s[64];
  __shared__ float rowsum_s[64];
  const int h = blockIdx.x, b = blockIdx.y;
  const int tid = threadIdx.x, lane = tid & 31, wid = tid >> 5;
  const int wm = wid >> 2, wn = wid & 3;
  const int hf = lane >> 4, idx = lane & 15;

  v16bf qf[2][2];
#pragma unroll
  for (int mi = 0; mi < 2; ++mi)
#pragma unroll
    for (int ks = 0; ks < 2; ++ks)
      qf[mi][ks] = load_frag_a(
          rep_bf + ((size_t)b * NQ + wm * 32 + mi * 16) * DIMC + h * DH + ks * 32,
          DIMC, lane);

  const u16* wbase = w_bf + (size_t)b * NTOK * DIMC + h * DH;

  // ---- pass 1: row maxima ----
  float rmax[16];
#pragma unroll
  for (int i = 0; i < 16; ++i) rmax[i] = -3.0e38f;
  for (int kb = 0; kb < 32; ++kb) {
    const int key0 = kb * 128 + wn * 32;
    v8f s[2][2] = {};
#pragma unroll
    for (int ks = 0; ks < 2; ++ks) {
      v16bf b0 = load_frag_b(wbase + (size_t)(key0)*DIMC + ks * 32, DIMC, lane);
      v16bf b1 = load_frag_b(wbase + (size_t)(key0 + 16) * DIMC + ks * 32, DIMC, lane);
      s[0][0] = wmma_bf16(qf[0][ks], b0, s[0][0]);
      s[0][1] = wmma_bf16(qf[0][ks], b1, s[0][1]);
      s[1][0] = wmma_bf16(qf[1][ks], b0, s[1][0]);
      s[1][1] = wmma_bf16(qf[1][ks], b1, s[1][1]);
    }
#pragma unroll
    for (int mi = 0; mi < 2; ++mi)
#pragma unroll
      for (int r = 0; r < 8; ++r)
        rmax[mi * 8 + r] = fmaxf(rmax[mi * 8 + r], fmaxf(s[mi][0][r], s[mi][1][r]));
  }
#pragma unroll
  for (int mi = 0; mi < 2; ++mi)
#pragma unroll
    for (int r = 0; r < 8; ++r) {
      float m = rmax[mi * 8 + r] * ATTN_SCALE;
#pragma unroll
      for (int msk = 1; msk < 16; msk <<= 1) m = fmaxf(m, __shfl_xor(m, msk, 32));
      if (idx == 0) redbuf[wn][wm * 32 + mi * 16 + hf * 8 + r] = m;
    }
  __syncthreads();
  if (tid < 64)
    rowmax_s[tid] = fmaxf(fmaxf(redbuf[0][tid], redbuf[1][tid]),
                          fmaxf(redbuf[2][tid], redbuf[3][tid]));

  // ---- pass 2 ----
  float rsum[16];
#pragma unroll
  for (int i = 0; i < 16; ++i) rsum[i] = 0.0f;
  v8f pv[2] = {};
  for (int kb = 0; kb < 32; ++kb) {
    __syncthreads();  // VT/PT reuse fence (also publishes rowmax_s first time)
    {   // stage V^T: 2 threads/key, 32 d-values each
      int key = tid >> 1, dloc = (tid & 1) * 32;
      const u16* src = wbase + (size_t)(kb * 128 + key) * DIMC + dloc;
#pragma unroll
      for (int c = 0; c < 4; ++c) {
        U4u u; u.v = *(const uint4*)(src + c * 8);
#pragma unroll
        for (int e = 0; e < 8; ++e) VT[(dloc + c * 8 + e) * 128 + key] = u.s[e];
      }
    }
    const int key0 = kb * 128 + wn * 32;
    v8f s[2][2] = {};
#pragma unroll
    for (int ks = 0; ks < 2; ++ks) {
      v16bf b0 = load_frag_b(wbase + (size_t)(key0)*DIMC + ks * 32, DIMC, lane);
      v16bf b1 = load_frag_b(wbase + (size_t)(key0 + 16) * DIMC + ks * 32, DIMC, lane);
      s[0][0] = wmma_bf16(qf[0][ks], b0, s[0][0]);
      s[0][1] = wmma_bf16(qf[0][ks], b1, s[0][1]);
      s[1][0] = wmma_bf16(qf[1][ks], b0, s[1][0]);
      s[1][1] = wmma_bf16(qf[1][ks], b1, s[1][1]);
    }
#pragma unroll
    for (int mi = 0; mi < 2; ++mi)
#pragma unroll
      for (int r = 0; r < 8; ++r) {
        int row = wm * 32 + mi * 16 + hf * 8 + r;
        float mx = rowmax_s[row];
        float p0 = __expf(s[mi][0][r] * ATTN_SCALE - mx);
        float p1 = __expf(s[mi][1][r] * ATTN_SCALE - mx);
        rsum[mi * 8 + r] += p0 + p1;
        PT[row * 128 + wn * 32 + idx]      = f2bf(p0);
        PT[row * 128 + wn * 32 + 16 + idx] = f2bf(p1);
      }
    __syncthreads();
    {   // vectorized unnormalized-attn store (consumers fold 1/rowsum)
      u16* dst = attn_bf + ((size_t)(b * HEADS + h) * NQ) * NTOK + kb * 128;
#pragma unroll
      for (int i = 0; i < 4; ++i) {
        int lin = tid + i * 256;                    // 1024 uint4 total
        int row = lin >> 4, c16 = lin & 15;
        *(uint4*)(dst + (size_t)row * NTOK + c16 * 8) =
            *(const uint4*)(PT + row * 128 + c16 * 8);
      }
    }
    // rep_delta partial: P(64x128) @ V(128x64); wave = rows wm*32.., cols wn*16..
#pragma unroll
    for (int ks = 0; ks < 4; ++ks) {
      v16bf bv = load_frag_b(VT + (wn * 16) * 128 + ks * 32, 128, lane);
#pragma unroll
      for (int mi = 0; mi < 2; ++mi) {
        v16bf ap = load_frag_a(PT + (wm * 32 + mi * 16) * 128 + ks * 32, 128, lane);
        pv[mi] = wmma_bf16(ap, bv, pv[mi]);
      }
    }
  }
  // row-sum reduction
#pragma unroll
  for (int mi = 0; mi < 2; ++mi)
#pragma unroll
    for (int r = 0; r < 8; ++r) {
      float sv = rsum[mi * 8 + r];
#pragma unroll
      for (int msk = 1; msk < 16; msk <<= 1) sv += __shfl_xor(sv, msk, 32);
      if (idx == 0) redbuf[wn][wm * 32 + mi * 16 + hf * 8 + r] = sv;
    }
  __syncthreads();
  if (tid < 64) {
    float total = redbuf[0][tid] + redbuf[1][tid] + redbuf[2][tid] + redbuf[3][tid];
    rowsum_s[tid] = total;
    rowsum_g[(b * HEADS + h) * NQ + tid] = total;
  }
  __syncthreads();
  float srep = step_rep[h];
#pragma unroll
  for (int mi = 0; mi < 2; ++mi)
#pragma unroll
    for (int r = 0; r < 8; ++r) {
      int row = wm * 32 + mi * 16 + hf * 8 + r;
      int d = wn * 16 + idx;
      float v = rep_f32[((size_t)b * NQ + row) * DIMC + h * DH + d] +
                srep * pv[mi][r] / rowsum_s[row];
      reph_bf[(((size_t)(b * HEADS + h)) * NQ + row) * DH + d] = f2bf(v);
    }
}

// ---------------------------------------------------------------------------
// K4: self-attention among pooled reps (64x64 per (b,h)), all-WMMA.
// Output x_delta2 stored TRANSPOSED ([d][q]) and pre-scaled by
// step_x[h] / rowsum1[b,h,q] so K5 is a plain GEMM.
// ---------------------------------------------------------------------------
__global__ __launch_bounds__(256) void k_attn2(
    const u16* __restrict__ reph_bf, const float* __restrict__ rowsum_g,
    const float* __restrict__ step_x, u16* __restrict__ xd2T) {
  __shared__ u16 R[64 * 64];
  __shared__ u16 RT[64 * 64];
  __shared__ float S2[64 * 64];
  __shared__ u16 P2[64 * 64];
  const int h = blockIdx.x, b = blockIdx.y;
  const int tid = threadIdx.x, lane = tid & 31, wid = tid >> 5;
  const int wm = wid >> 2, wn = wid & 3;
  const int hf = lane >> 4, idx = lane & 15;
  const u16* src = reph_bf + ((size_t)(b * HEADS + h)) * NQ * DH;
#pragma unroll
  for (int i = 0; i < 2; ++i) {                 // stage R and R^T
    int lin = tid * 2 + i;                      // 512 uint4
    int row = lin >> 3, c8 = lin & 7;
    U4u u; u.v = *(const uint4*)(src + row * 64 + c8 * 8);
    *(uint4*)(R + row * 64 + c8 * 8) = u.v;
#pragma unroll
    for (int e = 0; e < 8; ++e) RT[(c8 * 8 + e) * 64 + row] = u.s[e];
  }
  __syncthreads();
  v8f s2[2] = {};
#pragma unroll
  for (int ks = 0; ks < 2; ++ks) {
    v16bf bk = load_frag_b(R + (wn * 16) * 64 + ks * 32, 64, lane);
#pragma unroll
    for (int mi = 0; mi < 2; ++mi) {
      v16bf aq = load_frag_a(R + (wm * 32 + mi * 16) * 64 + ks * 32, 64, lane);
      s2[mi] = wmma_bf16(aq, bk, s2[mi]);
    }
  }
#pragma unroll
  for (int mi = 0; mi < 2; ++mi)
#pragma unroll
    for (int r = 0; r < 8; ++r)
      S2[(wm * 32 + mi * 16 + hf * 8 + r) * 64 + wn * 16 + idx] =
          s2[mi][r] * ATTN_SCALE;
  __syncthreads();
  if (tid < 64) {                               // normalized softmax rows
    float mx = -3.0e38f;
    for (int j = 0; j < 64; ++j) mx = fmaxf(mx, S2[tid * 64 + j]);
    float sum = 0.0f;
    for (int j = 0; j < 64; ++j) sum += __expf(S2[tid * 64 + j] - mx);
    float inv = 1.0f / sum;
    for (int j = 0; j < 64; ++j)
      P2[tid * 64 + j] = f2bf(__expf(S2[tid * 64 + j] - mx) * inv);
  }
  __syncthreads();
  v8f pv[2] = {};
#pragma unroll
  for (int ks = 0; ks < 2; ++ks) {
    v16bf bd = load_frag_b(RT + (wn * 16) * 64 + ks * 32, 64, lane);
#pragma unroll
    for (int mi = 0; mi < 2; ++mi) {
      v16bf ap = load_frag_a(P2 + (wm * 32 + mi * 16) * 64 + ks * 32, 64, lane);
      pv[mi] = wmma_bf16(ap, bd, pv[mi]);
    }
  }
  float sx = step_x[h];
#pragma unroll
  for (int mi = 0; mi < 2; ++mi)
#pragma unroll
    for (int r = 0; r < 8; ++r) {
      int qrow = wm * 32 + mi * 16 + hf * 8 + r;
      int d = wn * 16 + idx;
      float scl = sx / rowsum_g[(b * HEADS + h) * NQ + qrow];
      xd2T[(((size_t)(b * HEADS + h)) * DH + d) * NQ + qrow] = f2bf(pv[mi][r] * scl);
    }
}

// ---------------------------------------------------------------------------
// K5: upsample y[b,n,h*64+d] = sum_q attn[q,n] * xd2'[q,d]  (WMMA, attn^T
// staged through LDS per 128-row block; xd2T is already [d][q]).
// ---------------------------------------------------------------------------
__global__ __launch_bounds__(256) void k_upsample(
    const u16* __restrict__ attn_bf, const u16* __restrict__ xd2T,
    u16* __restrict__ y_bf) {
  __shared__ u16 AT[128 * 64];      // attn^T block [n][q]
  const int nb = blockIdx.x, h = blockIdx.y, b = blockIdx.z;
  const int tid = threadIdx.x, lane = tid & 31, wid = tid >> 5;
  const int wm = wid >> 1, wn = wid & 1;
  const int hf = lane >> 4, idx = lane & 15;
  {
    int q = tid >> 2, c0 = (tid & 3) * 32;
    const u16* src =
        attn_bf + ((size_t)(b * HEADS + h) * NQ + q) * NTOK + nb * 128 + c0;
#pragma unroll
    for (int c = 0; c < 4; ++c) {
      U4u u; u.v = *(const uint4*)(src + c * 8);
#pragma unroll
      for (int e = 0; e < 8; ++e) AT[(c0 + c * 8 + e) * 64 + q] = u.s[e];
    }
  }
  __syncthreads();
  const u16* bsrc = xd2T + ((size_t)(b * HEADS + h)) * DH * NQ;
  v8f acc[2][2] = {};
#pragma unroll
  for (int ks = 0; ks < 2; ++ks) {
    v16bf bf0 = load_frag_b(bsrc + (wn * 32) * 64 + ks * 32, 64, lane);
    v16bf bf1 = load_frag_b(bsrc + (wn * 32 + 16) * 64 + ks * 32, 64, lane);
#pragma unroll
    for (int mi = 0; mi < 2; ++mi) {
      v16bf af = load_frag_a(AT + (wm * 32 + mi * 16) * 64 + ks * 32, 64, lane);
      acc[mi][0] = wmma_bf16(af, bf0, acc[mi][0]);
      acc[mi][1] = wmma_bf16(af, bf1, acc[mi][1]);
    }
  }
#pragma unroll
  for (int mi = 0; mi < 2; ++mi)
#pragma unroll
    for (int ni = 0; ni < 2; ++ni)
#pragma unroll
      for (int r = 0; r < 8; ++r) {
        int n = nb * 128 + wm * 32 + mi * 16 + hf * 8 + r;
        int col = h * DH + wn * 32 + ni * 16 + idx;
        y_bf[((size_t)b * NTOK + n) * DIMC + col] = f2bf(acc[mi][ni][r]);
      }
}

// ---------------------------------------------------------------------------
extern "C" void kernel_launch(void* const* d_in, const int* in_sizes, int n_in,
                              void* d_out, int out_size, void* d_ws, size_t ws_size,
                              hipStream_t stream) {
  (void)in_sizes; (void)n_in; (void)out_size; (void)ws_size;
  const float* x    = (const float*)d_in[0];
  const float* pw   = (const float*)d_in[1];
  const float* srep = (const float*)d_in[2];
  const float* sx   = (const float*)d_in[3];
  const float* tw   = (const float*)d_in[4];
  const float* tb   = (const float*)d_in[5];
  float* out = (float*)d_out;

  const size_t NX   = (size_t)BATCH * NTOK * DIMC;      // 16,777,216
  const size_t NW   = (size_t)DIMC * DIMC;              // 262,144
  const size_t NREP = (size_t)BATCH * NQ * DIMC;        // 262,144
  const size_t NRH  = (size_t)BATCH * HEADS * NQ * DH;  // 262,144
  const size_t NRS  = (size_t)BATCH * HEADS * NQ;       // 4,096

  char* p = (char*)d_ws;                 // ~104 MB total
  u16* xy_bf   = (u16*)p; p += NX * 2;   // x (bf16), later reused for y
  u16* w_bf    = (u16*)p; p += NX * 2;   // proj output tokens (K/V)
  u16* attn_bf = (u16*)p; p += NX * 2;   // unnormalized attention
  u16* pw_bf   = (u16*)p; p += NW * 2;
  u16* tw_bf   = (u16*)p; p += NW * 2;
  u16* rep_bf  = (u16*)p; p += NREP * 2;
  u16* reph_bf = (u16*)p; p += NRH * 2;
  u16* xd2T    = (u16*)p; p += NRH * 2;
  float* rep_f32 = (float*)p; p += NREP * 4;
  float* rowsum  = (float*)p; p += NRS * 4;

  const int cvt_total = (int)(NX + 2 * NW);
  k_convert<<<dim3((cvt_total + 255) / 256), dim3(256), 0, stream>>>(
      x, pw, tw, xy_bf, pw_bf, tw_bf);
  k_gemm<<<dim3(256, 8), dim3(256), 0, stream>>>(
      xy_bf, pw_bf, (const float*)nullptr, w_bf, (float*)nullptr);
  k_pool<<<dim3(128), dim3(256), 0, stream>>>(w_bf, rep_f32, rep_bf);
  k_attn1<<<dim3(HEADS, BATCH), dim3(256), 0, stream>>>(
      w_bf, rep_bf, rep_f32, srep, attn_bf, reph_bf, rowsum);
  k_attn2<<<dim3(HEADS, BATCH), dim3(256), 0, stream>>>(
      reph_bf, rowsum, sx, xd2T);
  k_upsample<<<dim3(32, HEADS, BATCH), dim3(256), 0, stream>>>(
      attn_bf, xd2T, xy_bf /* y reuses x's buffer */);
  k_gemm<<<dim3(256, 8), dim3(256), 0, stream>>>(
      xy_bf, tw_bf, tb, (u16*)nullptr, out);
}
